// SelfAttention_3083786518544
// MI455X (gfx1250) — matrix operand order repaired and checked by reading, not compiled, not run on hardware
//
#include <hip/hip_runtime.h>

#define D_MODEL 1024
#define N_HEADS 16
#define D_HEAD  64
#define BATCH   2
#define SEQ     2048
#define ROWS    (BATCH*SEQ)   // 4096

typedef __attribute__((ext_vector_type(16))) _Float16 v16h;
typedef __attribute__((ext_vector_type(8)))  _Float16 v8h;
typedef __attribute__((ext_vector_type(8)))  float    v8f;

union V16 { v16h v; v8h h8[2]; };

// A-operand fragment of a 16x32 f16 tile (row-major, K contiguous).
// lane (lo,hi): row = lo; halves j=0..7 -> K=kb+8*hi+j ; j=8..15 -> K=kb+16+8*hi+(j-8)
__device__ __forceinline__ V16 load_a(const _Float16* __restrict__ row, int kb, int hi) {
  V16 f;
  f.h8[0] = *(const v8h*)(row + kb + 8 * hi);
  f.h8[1] = *(const v8h*)(row + kb + 16 + 8 * hi);
  return f;
}
// B-operand fragment of a 32x16 f16 tile, accessed via Bt (column-major B == K contiguous).
// lane (lo,hi): col = lo; halves j=0..7 -> K=kb+8*hi+j ; j=8..15 -> K=kb+16+8*hi+(j-8)
__device__ __forceinline__ V16 load_b(const _Float16* __restrict__ col, int kb, int hi) {
  V16 f;
  f.h8[0] = *(const v8h*)(col + kb + 8 * hi);
  f.h8[1] = *(const v8h*)(col + kb + 16 + 8 * hi);
  return f;
}

__device__ __forceinline__ v8f wmma32(const V16& a, const V16& b, v8f c) {
  return __builtin_amdgcn_wmma_f32_16x16x32_f16(false, a.v, false, b.v,
                                                (short)0, c, false, false);
}

// ---------------- pack kernels ----------------
__global__ void pack_f32_to_f16(const float* __restrict__ in, _Float16* __restrict__ out, int n) {
  int i = blockIdx.x * blockDim.x + threadIdx.x;
  if (i < n) out[i] = (_Float16)in[i];
}

// W[h][d][e] (16,1024,64) -> Wt[h][e][d] f16
__global__ void pack_w_transpose(const float* __restrict__ w, _Float16* __restrict__ wt) {
  int i = blockIdx.x * blockDim.x + threadIdx.x;   // 0 .. 16*64*1024-1
  int h = i >> 16;
  int rem = i & 65535;
  int e = rem >> 10;
  int d = rem & 1023;
  wt[i] = (_Float16)w[(h << 16) + d * D_HEAD + e];
}

// W_O flat is [he][d] (1024x1024) -> WOt[d][he] f16
__global__ void pack_wo_transpose(const float* __restrict__ w, _Float16* __restrict__ wt) {
  int i = blockIdx.x * blockDim.x + threadIdx.x;   // 0 .. 1048575
  int d = i >> 10;
  int he = i & 1023;
  wt[i] = (_Float16)w[he * D_MODEL + d];
}

// ---------------- QKV projection GEMM ----------------
// One wave: 32 s-rows x 64 e-cols for one (b,h).  C = (X[s,1024] * W[1024,64] + bias) * out_scale
__global__ void qkv_gemm(const _Float16* __restrict__ X,     // [ROWS][1024]
                         const _Float16* __restrict__ Wt,    // [H][64][1024] (e-major, d contiguous)
                         const float* __restrict__ bias,     // [H][64]
                         _Float16* __restrict__ out,         // rowmajor: [b][h][s][64]; transposed: [b][h][64][s]
                         int transpose_out, float out_scale) {
  int lane = threadIdx.x & 31;
  int wid  = threadIdx.x >> 5;
  int lo = lane & 15, hi = lane >> 4;
  int s_tile = blockIdx.x * 4 + wid;            // 0..63, 32 rows each
  int bh = blockIdx.y;                          // 0..31
  int b = bh >> 4, h = bh & 15;
  int s_base = s_tile * 32;

  const _Float16* Arow0 = X + (size_t)(b * SEQ + s_base + lo) * D_MODEL;
  const _Float16* Arow1 = Arow0 + (size_t)16 * D_MODEL;
  const _Float16* Bbase = Wt + (size_t)h * D_HEAD * D_MODEL;

  v8f zero = {};
  v8f c[2][4];
#pragma unroll
  for (int u = 0; u < 2; ++u)
#pragma unroll
    for (int t = 0; t < 4; ++t) c[u][t] = zero;

  for (int kk = 0; kk < D_MODEL; kk += 32) {
    V16 a0 = load_a(Arow0, kk, hi);
    V16 a1 = load_a(Arow1, kk, hi);
#pragma unroll
    for (int t = 0; t < 4; ++t) {
      V16 bf = load_b(Bbase + (size_t)(16 * t + lo) * D_MODEL, kk, hi);
      c[0][t] = wmma32(a0, bf, c[0][t]);
      c[1][t] = wmma32(a1, bf, c[1][t]);
    }
  }
#pragma unroll
  for (int u = 0; u < 2; ++u) {
#pragma unroll
    for (int t = 0; t < 4; ++t) {
      int e = 16 * t + lo;
      float bv = bias[h * D_HEAD + e];
#pragma unroll
      for (int r = 0; r < 8; ++r) {
        int m = 16 * u + r + 8 * hi;
        float val = (c[u][t][r] + bv) * out_scale;
        if (!transpose_out)
          out[((size_t)(b * N_HEADS + h) * SEQ + s_base + m) * D_HEAD + e] = (_Float16)val;
        else
          out[((size_t)(b * N_HEADS + h) * D_HEAD + e) * SEQ + s_base + m] = (_Float16)val;
      }
    }
  }
}

// ---------------- flash attention ----------------
// One wave per (b,h,16-row q tile); streams 32 k-positions per iteration.
// Q is pre-scaled by 1/sqrt(d_head).  Row sums accumulate on the matrix pipe
// via zl += P @ ones, so each lane ends up holding l[m] for exactly its rows.
__global__ void attn_kernel(const _Float16* __restrict__ Q,   // [b][h][s][64], pre-scaled
                            const _Float16* __restrict__ K,   // [b][h][s][64]
                            const _Float16* __restrict__ Vt,  // [b][h][64][s]
                            _Float16* __restrict__ Z) {       // [b][s][h][64]
  __shared__ alignas(16) _Float16 Pld[4][16][32];
  int lane = threadIdx.x & 31;
  int wid  = threadIdx.x >> 5;
  int lo = lane & 15, hi = lane >> 4;
  int q_tile = blockIdx.x * 4 + wid;            // 0..127
  int bh = blockIdx.y;
  int b = bh >> 4, h = bh & 15;
  int q_base = q_tile * 16;

  const _Float16* Qb = Q  + (size_t)(b * N_HEADS + h) * SEQ * D_HEAD;
  const _Float16* Kb = K  + (size_t)(b * N_HEADS + h) * SEQ * D_HEAD;
  const _Float16* Vb = Vt + (size_t)(b * N_HEADS + h) * D_HEAD * SEQ;

  V16 qf0 = load_a(Qb + (size_t)(q_base + lo) * D_HEAD, 0, hi);
  V16 qf1 = load_a(Qb + (size_t)(q_base + lo) * D_HEAD, 32, hi);

  V16 onesf;
#pragma unroll
  for (int i = 0; i < 16; ++i) onesf.v[i] = (_Float16)1.0f;

  v8f zero = {};
  v8f zc[4] = {zero, zero, zero, zero};
  v8f zl = zero;                      // row-sum accumulator (all columns identical)
  float mrow[8];
#pragma unroll
  for (int r = 0; r < 8; ++r) mrow[r] = -1e30f;

  int q_hi = q_base + 15;
  for (int kb = 0; kb <= q_hi; kb += 32) {
    // two 16x16 score tiles, K-dim = d_head (2 WMMA steps of 32 each)
    v8f s0 = zero, s1 = zero;
    {
      const _Float16* k0 = Kb + (size_t)(kb + lo) * D_HEAD;
      s0 = wmma32(qf0, load_b(k0, 0, hi), s0);
      s0 = wmma32(qf1, load_b(k0, 32, hi), s0);
      const _Float16* k1 = Kb + (size_t)(kb + 16 + lo) * D_HEAD;
      s1 = wmma32(qf0, load_b(k1, 0, hi), s1);
      s1 = wmma32(qf1, load_b(k1, 32, hi), s1);
    }
    float p0[8], p1[8];
#pragma unroll
    for (int r = 0; r < 8; ++r) {
      int m_g = q_base + r + 8 * hi;
      p0[r] = ((kb + lo)      <= m_g) ? s0[r] : -1e30f;
      p1[r] = ((kb + 16 + lo) <= m_g) ? s1[r] : -1e30f;
    }
    // online softmax: only the row-max needs cross-lane traffic
#pragma unroll
    for (int r = 0; r < 8; ++r) {
      float rm = fmaxf(p0[r], p1[r]);
#pragma unroll
      for (int off = 1; off < 16; off <<= 1)
        rm = fmaxf(rm, __shfl_xor(rm, off, 16));
      float mnew  = fmaxf(mrow[r], rm);
      float alpha = __expf(mrow[r] - mnew);
      p0[r] = __expf(p0[r] - mnew);
      p1[r] = __expf(p1[r] - mnew);
      mrow[r] = mnew;
      zl[r] *= alpha;
#pragma unroll
      for (int t = 0; t < 4; ++t) zc[t][r] *= alpha;
    }
    // C-layout -> A-layout for P via LDS (same-wave LDS ops are in-order)
#pragma unroll
    for (int r = 0; r < 8; ++r) {
      Pld[wid][r + 8 * hi][lo]      = (_Float16)p0[r];
      Pld[wid][r + 8 * hi][16 + lo] = (_Float16)p1[r];
    }
    asm volatile("s_wait_dscnt 0" ::: "memory");
    V16 pa;
    pa.h8[0] = *(const v8h*)(&Pld[wid][lo][8 * hi]);
    pa.h8[1] = *(const v8h*)(&Pld[wid][lo][16 + 8 * hi]);
    // l += rowsum(P) on the matrix pipe; Z += P(16x32) @ V(32x64)
    zl = wmma32(pa, onesf, zl);
#pragma unroll
    for (int t = 0; t < 4; ++t) {
      const _Float16* vcol = Vb + (size_t)(16 * t + lo) * SEQ;
      zc[t] = wmma32(pa, load_b(vcol, kb, hi), zc[t]);
    }
  }
  // normalize and store Z[b][s][h][e] (f16, feeds output GEMM as A operand)
#pragma unroll
  for (int t = 0; t < 4; ++t) {
    int e = 16 * t + lo;
#pragma unroll
    for (int r = 0; r < 8; ++r) {
      int s = q_base + r + 8 * hi;
      float val = zc[t][r] / zl[r];
      Z[((size_t)(b * SEQ + s) * N_HEADS + h) * D_HEAD + e] = (_Float16)val;
    }
  }
}

// ---------------- output projection GEMM ----------------
// One wave: 32 s-rows x 64 d-cols.  out[s][d] = Z[s][he] * WO[he][d] + bO[d], f32 output
__global__ void out_gemm(const _Float16* __restrict__ Zb,   // [ROWS][1024]
                         const _Float16* __restrict__ WOt,  // [1024 d][1024 he]
                         const float* __restrict__ bO,      // [1024]
                         float* __restrict__ out) {         // [ROWS][1024]
  int lane = threadIdx.x & 31;
  int wid  = threadIdx.x >> 5;
  int lo = lane & 15, hi = lane >> 4;
  int s_base = (blockIdx.x * 4 + wid) * 32;   // 128 row-tiles of 32
  int d_base = blockIdx.y * 64;               // 16 d-groups

  const _Float16* Arow0 = Zb + (size_t)(s_base + lo) * D_MODEL;
  const _Float16* Arow1 = Arow0 + (size_t)16 * D_MODEL;
  v8f zero = {};
  v8f c[2][4];
#pragma unroll
  for (int u = 0; u < 2; ++u)
#pragma unroll
    for (int t = 0; t < 4; ++t) c[u][t] = zero;

  for (int kk = 0; kk < D_MODEL; kk += 32) {
    V16 a0 = load_a(Arow0, kk, hi);
    V16 a1 = load_a(Arow1, kk, hi);
#pragma unroll
    for (int t = 0; t < 4; ++t) {
      V16 bf = load_b(WOt + (size_t)(d_base + 16 * t + lo) * D_MODEL, kk, hi);
      c[0][t] = wmma32(a0, bf, c[0][t]);
      c[1][t] = wmma32(a1, bf, c[1][t]);
    }
  }
#pragma unroll
  for (int u = 0; u < 2; ++u) {
#pragma unroll
    for (int t = 0; t < 4; ++t) {
      int d = d_base + 16 * t + lo;
      float bv = bO[d];
#pragma unroll
      for (int r = 0; r < 8; ++r) {
        int s = s_base + 16 * u + r + 8 * hi;
        out[(size_t)s * D_MODEL + d] = c[u][t][r] + bv;
      }
    }
  }
}

extern "C" void kernel_launch(void* const* d_in, const int* in_sizes, int n_in,
                              void* d_out, int out_size, void* d_ws, size_t ws_size,
                              hipStream_t stream) {
  const float* resid = (const float*)d_in[0];
  const float* W_Q = (const float*)d_in[1];
  const float* b_Q = (const float*)d_in[2];
  const float* W_K = (const float*)d_in[3];
  const float* b_K = (const float*)d_in[4];
  const float* W_V = (const float*)d_in[5];
  const float* b_V = (const float*)d_in[6];
  const float* W_O = (const float*)d_in[7];
  const float* b_O = (const float*)d_in[8];
  float* out = (float*)d_out;

  char* ws = (char*)d_ws;                         // 48 MB used
  _Float16* X16  = (_Float16*)(ws);               // [4096][1024]      8 MB
  _Float16* Q16  = (_Float16*)(ws + (8u  << 20)); // [b][h][s][64]     8 MB
  _Float16* K16  = (_Float16*)(ws + (16u << 20)); // [b][h][s][64]     8 MB
  _Float16* Vt16 = (_Float16*)(ws + (24u << 20)); // [b][h][64][s]     8 MB
  _Float16* Zb   = (_Float16*)(ws + (32u << 20)); // [b][s][h][64]     8 MB
  _Float16* WQt  = (_Float16*)(ws + (40u << 20)); // [h][64][1024]     2 MB
  _Float16* WKt  = (_Float16*)(ws + (42u << 20)); //                   2 MB
  _Float16* WVt  = (_Float16*)(ws + (44u << 20)); //                   2 MB
  _Float16* WOt  = (_Float16*)(ws + (46u << 20)); // [1024][1024]      2 MB

  pack_f32_to_f16<<<dim3(ROWS * D_MODEL / 256), 256, 0, stream>>>(resid, X16, ROWS * D_MODEL);
  pack_w_transpose<<<dim3(4096), 256, 0, stream>>>(W_Q, WQt);
  pack_w_transpose<<<dim3(4096), 256, 0, stream>>>(W_K, WKt);
  pack_w_transpose<<<dim3(4096), 256, 0, stream>>>(W_V, WVt);
  pack_wo_transpose<<<dim3(4096), 256, 0, stream>>>(W_O, WOt);

  const float qscale = 0.125f;   // 1/sqrt(D_HEAD), folded into Q
  qkv_gemm<<<dim3(16, 32), 128, 0, stream>>>(X16, WQt, b_Q, Q16, 0, qscale);
  qkv_gemm<<<dim3(16, 32), 128, 0, stream>>>(X16, WKt, b_K, K16, 0, 1.0f);
  qkv_gemm<<<dim3(16, 32), 128, 0, stream>>>(X16, WVt, b_V, Vt16, 1, 1.0f);

  attn_kernel<<<dim3(32, 32), 128, 0, stream>>>(Q16, K16, Vt16, Zb);

  out_gemm<<<dim3(32, 16), 128, 0, stream>>>(Zb, WOt, b_O, out);
}
